// MambaSSMBlock_63660005261822
// MI455X (gfx1250) — matrix operand order repaired and checked by reading, not compiled, not run on hardware
//
#include <hip/hip_runtime.h>

typedef __bf16 bf16_t;
typedef __attribute__((ext_vector_type(16))) __bf16 v16bf;
typedef __attribute__((ext_vector_type(8)))  __bf16 v8bf;
typedef __attribute__((ext_vector_type(8)))  float  v8f;

#define D_MODEL 1024
#define D_STATE 16
#define D_CONV  4
#define D_INNER 2048
#define B_SZ    2
#define L_SEQ   2048
#define M_ROWS  (B_SZ * L_SEQ)     // 4096 rows of (b,l)
#define XZ_N    (2 * D_INNER)      // 4096
#define XP_N    (2 * D_STATE + 1)  // 33

__device__ __forceinline__ bf16_t f2bf(float f) {
  unsigned u = __float_as_uint(f);
  u += 0x7FFFu + ((u >> 16) & 1u);           // round-to-nearest-even
  unsigned short h = (unsigned short)(u >> 16);
  return __builtin_bit_cast(bf16_t, h);
}

__device__ __forceinline__ float sigmoidf_(float v) {
  return 1.0f / (1.0f + expf(-v));
}

// ---------------------------------------------------------------------------
// 1) LayerNorm over D_MODEL, fused cast to bf16. One block per (b,l) row.
// ---------------------------------------------------------------------------
__global__ void ln_cast_kernel(const float* __restrict__ x,
                               const float* __restrict__ gamma,
                               const float* __restrict__ beta,
                               bf16_t* __restrict__ xn) {
  const int m = blockIdx.x;
  const float* row = x + (size_t)m * D_MODEL;
  __shared__ float red[256];
  float s = 0.f, s2 = 0.f;
  for (int i = threadIdx.x; i < D_MODEL; i += blockDim.x) {
    float v = row[i];
    s += v; s2 += v * v;
  }
  red[threadIdx.x] = s; __syncthreads();
  for (int off = 128; off > 0; off >>= 1) {
    if (threadIdx.x < off) red[threadIdx.x] += red[threadIdx.x + off];
    __syncthreads();
  }
  float mean = red[0] * (1.0f / D_MODEL);
  __syncthreads();
  red[threadIdx.x] = s2; __syncthreads();
  for (int off = 128; off > 0; off >>= 1) {
    if (threadIdx.x < off) red[threadIdx.x] += red[threadIdx.x + off];
    __syncthreads();
  }
  float var = red[0] * (1.0f / D_MODEL) - mean * mean;
  float inv = rsqrtf(var + 1e-5f);
  for (int i = threadIdx.x; i < D_MODEL; i += blockDim.x) {
    float v = (row[i] - mean) * inv * gamma[i] + beta[i];
    xn[(size_t)m * D_MODEL + i] = f2bf(v);
  }
}

// ---------------------------------------------------------------------------
// 2) f32 -> bf16 weight cast
// ---------------------------------------------------------------------------
__global__ void cast_f32_bf16(const float* __restrict__ in,
                              bf16_t* __restrict__ out, int n) {
  int i = blockIdx.x * blockDim.x + threadIdx.x;
  if (i < n) out[i] = f2bf(in[i]);
}

// ---------------------------------------------------------------------------
// 3) WMMA GEMM: C[M,N] = A[M,K] * B[N,K]^T  (bf16 inputs, f32 out)
//    Wave tile 32(M) x 64(N): 2 A-fragments, 4 B-fragments, 8 accumulators.
//    launch_bounds(256,1) relaxes the VGPR budget so nothing spills.
//    Wave-uniform base pointers + 32-bit lane offsets -> GVS addressing,
//    no 64-bit vector adds in the hot loop.
//    K mult of 32; M mult of 32; N mult of 64.
// ---------------------------------------------------------------------------
__global__ __launch_bounds__(256, 1)
void wmma_gemm_abT(const bf16_t* __restrict__ A,
                   const bf16_t* __restrict__ Bm,
                   float* __restrict__ C,
                   int M, int N, int K) {
  const int lane  = threadIdx.x & 31;
  const int gwave = (blockIdx.x * blockDim.x + threadIdx.x) >> 5;
  const int tilesN = N >> 6;
  const int tm = gwave / tilesN;    // 32-row tile index
  const int tn = gwave % tilesN;    // 64-col tile index
  if (tm * 32 >= M) return;
  const int half = lane >> 4;       // 0: lanes 0-15, 1: lanes 16-31
  const int l16  = lane & 15;

  // Wave-uniform bases (SGPR), per-lane 32-bit byte offsets (VGPR).
  const char* Ab = (const char*)(A  + (size_t)tm * 32 * K);
  const char* Bb = (const char*)(Bm + (size_t)tn * 64 * K);
  const unsigned rowBytes = (unsigned)K * 2u;
  const unsigned aoff0 = (unsigned)l16 * rowBytes + (unsigned)half * 32u;
  const unsigned aoff1 = aoff0 + 16u * rowBytes;
  const unsigned boff0 = (unsigned)l16 * rowBytes + (unsigned)half * 32u;
  const unsigned boff1 = boff0 + 16u * rowBytes;
  const unsigned boff2 = boff0 + 32u * rowBytes;
  const unsigned boff3 = boff0 + 48u * rowBytes;

  v8f acc[8] = {};   // [mi*4 + s]

#pragma unroll 2
  for (int k0 = 0; k0 < K; k0 += 32) {
    const unsigned kb = (unsigned)k0 * 2u;
    // A fragments (16x32 bf16): elems 0..7 -> K=k0+half*16+j,
    // elems 8..15 -> K=k0+half*16+8+j  (second chunk = +16 bytes)
    union { v16bf v; v8bf h[2]; } a0, a1;
    a0.h[0] = *(const v8bf*)(Ab + aoff0 + kb);
    a0.h[1] = *(const v8bf*)(Ab + aoff0 + kb + 16u);
    a1.h[0] = *(const v8bf*)(Ab + aoff1 + kb);
    a1.h[1] = *(const v8bf*)(Ab + aoff1 + kb + 16u);
    // B fragments (32x16 bf16): lane holds column, elems = K k0+half*16..+15
    v16bf b0 = *(const v16bf*)(Bb + boff0 + kb);
    v16bf b1 = *(const v16bf*)(Bb + boff1 + kb);
    v16bf b2 = *(const v16bf*)(Bb + boff2 + kb);
    v16bf b3 = *(const v16bf*)(Bb + boff3 + kb);

    if (k0 + 32 < K) {                        // global_prefetch_b8 next K-slab
      __builtin_prefetch(Ab + aoff0 + kb + 64u, 0, 1);
      __builtin_prefetch(Ab + aoff1 + kb + 64u, 0, 1);
      __builtin_prefetch(Bb + boff0 + kb + 64u, 0, 1);
      __builtin_prefetch(Bb + boff2 + kb + 64u, 0, 1);
    }

    acc[0] = __builtin_amdgcn_wmma_f32_16x16x32_bf16(false, a0.v, false, b0,
                                                     (short)0, acc[0], false, false);
    acc[1] = __builtin_amdgcn_wmma_f32_16x16x32_bf16(false, a0.v, false, b1,
                                                     (short)0, acc[1], false, false);
    acc[2] = __builtin_amdgcn_wmma_f32_16x16x32_bf16(false, a0.v, false, b2,
                                                     (short)0, acc[2], false, false);
    acc[3] = __builtin_amdgcn_wmma_f32_16x16x32_bf16(false, a0.v, false, b3,
                                                     (short)0, acc[3], false, false);
    acc[4] = __builtin_amdgcn_wmma_f32_16x16x32_bf16(false, a1.v, false, b0,
                                                     (short)0, acc[4], false, false);
    acc[5] = __builtin_amdgcn_wmma_f32_16x16x32_bf16(false, a1.v, false, b1,
                                                     (short)0, acc[5], false, false);
    acc[6] = __builtin_amdgcn_wmma_f32_16x16x32_bf16(false, a1.v, false, b2,
                                                     (short)0, acc[6], false, false);
    acc[7] = __builtin_amdgcn_wmma_f32_16x16x32_bf16(false, a1.v, false, b3,
                                                     (short)0, acc[7], false, false);
  }

  // C/D layout: lane l -> col l16, VGPR v -> row v + 8*half (per 16x16 tile)
  float* Cb = C + (size_t)tm * 32 * N + tn * 64;
#pragma unroll
  for (int mi = 0; mi < 2; ++mi)
#pragma unroll
    for (int s = 0; s < 4; ++s)
#pragma unroll
      for (int v = 0; v < 8; ++v)
        Cb[(unsigned)(mi * 16 + half * 8 + v) * (unsigned)N + s * 16 + l16] =
            acc[mi * 4 + s][v];
}

// ---------------------------------------------------------------------------
// 4) Depthwise causal conv (K=4) + bias + SiLU. x_ssm is the low half of xz.
// ---------------------------------------------------------------------------
__global__ void conv_silu_kernel(const float* __restrict__ xz,
                                 const float* __restrict__ cw,
                                 const float* __restrict__ cb,
                                 float* __restrict__ xc) {
  int idx = blockIdx.x * blockDim.x + threadIdx.x;
  if (idx >= B_SZ * L_SEQ * D_INNER) return;
  int d = idx % D_INNER;
  int l = (idx / D_INNER) % L_SEQ;
  int b = idx / (D_INNER * L_SEQ);
  float acc = cb[d];
#pragma unroll
  for (int k = 0; k < D_CONV; ++k) {
    int ls = l - (D_CONV - 1) + k;
    if (ls >= 0)
      acc += cw[d * D_CONV + k] * xz[((size_t)(b * L_SEQ + ls)) * XZ_N + d];
  }
  xc[(size_t)idx] = acc * sigmoidf_(acc);
}

// ---------------------------------------------------------------------------
// 5) xp[m][j] = sum_d xc[m][d] * W_xproj[j][d]   (N=33, plain VALU)
// ---------------------------------------------------------------------------
__global__ void xproj_kernel(const float* __restrict__ xc,
                             const float* __restrict__ Wx,
                             float* __restrict__ xp) {
  int t = blockIdx.x * blockDim.x + threadIdx.x;
  if (t >= M_ROWS * XP_N) return;
  int m = t / XP_N, j = t % XP_N;
  const float* xr = xc + (size_t)m * D_INNER;
  const float* wr = Wx + (size_t)j * D_INNER;
  float s = 0.f;
  for (int d = 0; d < D_INNER; ++d) s += xr[d] * wr[d];
  xp[t] = s;
}

// ---------------------------------------------------------------------------
// 6) dt[m][d] = softplus(xp[m] . W_dt[d] + b_dt[d]) + 0.001
// ---------------------------------------------------------------------------
__global__ void dt_kernel(const float* __restrict__ xp,
                          const float* __restrict__ Wdt,
                          const float* __restrict__ bdt,
                          float* __restrict__ dt) {
  int t = blockIdx.x * blockDim.x + threadIdx.x;
  if (t >= M_ROWS * D_INNER) return;
  int m = t / D_INNER, d = t % D_INNER;
  const float* xr = xp + (size_t)m * XP_N;
  const float* wr = Wdt + (size_t)d * XP_N;
  float s = bdt[d];
#pragma unroll
  for (int j = 0; j < XP_N; ++j) s += xr[j] * wr[j];
  float sp = (s > 20.f) ? s : log1pf(expf(s));
  dt[t] = sp + 0.001f;
}

// ---------------------------------------------------------------------------
// 7) Sequential SSM scan: one thread per (b,d), 16 states in registers.
//    Writes ys strided into the (dead) x_ssm half of xz: ys[m,d] = xz[m*XZ_N+d]
// ---------------------------------------------------------------------------
__global__ void scan_kernel(const float* __restrict__ xc,
                            const float* __restrict__ dt,
                            const float* __restrict__ xp,
                            const float* __restrict__ A_log,
                            const float* __restrict__ Dp,
                            float* __restrict__ ys /* stride XZ_N */) {
  int t = blockIdx.x * blockDim.x + threadIdx.x;
  if (t >= B_SZ * D_INNER) return;
  int b = t / D_INNER, d = t % D_INNER;
  float A[D_STATE], h[D_STATE];
#pragma unroll
  for (int n = 0; n < D_STATE; ++n) {
    A[n] = -expf(A_log[d * D_STATE + n]);
    h[n] = 0.f;
  }
  float Dpar = Dp[d];
  for (int l = 0; l < L_SEQ; ++l) {
    size_t m = (size_t)b * L_SEQ + l;
    float xt  = xc[m * D_INNER + d];
    float dtt = dt[m * D_INNER + d];
    const float* xr = xp + m * XP_N;
    float y = Dpar * xt;
#pragma unroll
    for (int n = 0; n < D_STATE; ++n) {
      float dA = expf(A[n] * dtt);
      float dB = (1.f - dA) / (A[n] + 1e-8f) * xr[1 + n];
      float hn = dA * h[n] + dB * xt;
      hn = fminf(10.f, fmaxf(-10.f, hn));
      h[n] = hn;
      y += xr[1 + D_STATE + n] * hn;
    }
    ys[m * XZ_N + d] = y;
  }
}

// ---------------------------------------------------------------------------
// 8) Gate: yg[m][d] = bf16( ys[m][d] * silu(z[m][d]) );  ys in low half of xz,
//    z in high half.
// ---------------------------------------------------------------------------
__global__ void gate_cast_kernel(const float* __restrict__ xz,
                                 bf16_t* __restrict__ yg) {
  int t = blockIdx.x * blockDim.x + threadIdx.x;
  if (t >= M_ROWS * D_INNER) return;
  int m = t / D_INNER, d = t % D_INNER;
  float ysv = xz[(size_t)m * XZ_N + d];
  float z   = xz[(size_t)m * XZ_N + D_INNER + d];
  yg[t] = f2bf(ysv * (z * sigmoidf_(z)));
}

// ---------------------------------------------------------------------------
// Launch
// ---------------------------------------------------------------------------
extern "C" void kernel_launch(void* const* d_in, const int* in_sizes, int n_in,
                              void* d_out, int out_size, void* d_ws, size_t ws_size,
                              hipStream_t stream) {
  (void)in_sizes; (void)n_in; (void)out_size; (void)ws_size;
  const float* x       = (const float*)d_in[0];
  const float* ln_g    = (const float*)d_in[1];
  const float* ln_b    = (const float*)d_in[2];
  const float* W_in    = (const float*)d_in[3];
  const float* conv_w  = (const float*)d_in[4];
  const float* conv_b  = (const float*)d_in[5];
  const float* A_log   = (const float*)d_in[6];
  const float* D_param = (const float*)d_in[7];
  const float* W_xproj = (const float*)d_in[8];
  const float* W_dt    = (const float*)d_in[9];
  const float* b_dt    = (const float*)d_in[10];
  const float* W_out   = (const float*)d_in[11];

  char* ws = (char*)d_ws;
  size_t off = 0;
  auto carve = [&](size_t bytes) -> char* {
    char* p = ws + off;
    off += (bytes + 255) & ~(size_t)255;
    return p;
  };
  bf16_t* xn_bf   = (bf16_t*)carve((size_t)M_ROWS * D_MODEL * 2);  //  8 MB \ reused as
  bf16_t* Win_bf  = (bf16_t*)carve((size_t)XZ_N   * D_MODEL * 2);  //  8 MB / yg (16 MB)
  float*  xz      = (float*) carve((size_t)M_ROWS * XZ_N    * 4);  // 64 MB
  float*  xc      = (float*) carve((size_t)M_ROWS * D_INNER * 4);  // 32 MB
  float*  xp      = (float*) carve((size_t)M_ROWS * XP_N    * 4);  // 0.5 MB
  float*  dtb     = (float*) carve((size_t)M_ROWS * D_INNER * 4);  // 32 MB
  bf16_t* Wout_bf = (bf16_t*)carve((size_t)D_MODEL * D_INNER * 2); //  4 MB
  bf16_t* yg      = xn_bf;  // region A reused after GEMM1 consumers finish

  // 1) LayerNorm + cast
  ln_cast_kernel<<<M_ROWS, 256, 0, stream>>>(x, ln_g, ln_b, xn_bf);

  // 2) weight casts
  {
    int n1 = XZ_N * D_MODEL;
    cast_f32_bf16<<<(n1 + 255) / 256, 256, 0, stream>>>(W_in, Win_bf, n1);
    int n2 = D_MODEL * D_INNER;
    cast_f32_bf16<<<(n2 + 255) / 256, 256, 0, stream>>>(W_out, Wout_bf, n2);
  }

  // 3) xz = xn @ W_in^T   (M=4096, N=4096, K=1024)
  {
    int waves = (M_ROWS / 32) * (XZ_N / 64);   // 128 * 64 = 8192 waves
    wmma_gemm_abT<<<(waves * 32) / 256, 256, 0, stream>>>(
        xn_bf, Win_bf, xz, M_ROWS, XZ_N, D_MODEL);
  }

  // 4) depthwise conv + SiLU
  {
    int tot = B_SZ * L_SEQ * D_INNER;
    conv_silu_kernel<<<(tot + 255) / 256, 256, 0, stream>>>(xz, conv_w, conv_b, xc);
  }

  // 5) xp = xc @ W_xproj^T
  {
    int tot = M_ROWS * XP_N;
    xproj_kernel<<<(tot + 255) / 256, 256, 0, stream>>>(xc, W_xproj, xp);
  }

  // 6) dt
  {
    int tot = M_ROWS * D_INNER;
    dt_kernel<<<(tot + 255) / 256, 256, 0, stream>>>(xp, W_dt, b_dt, dtb);
  }

  // 7) scan (ys written strided into low half of xz)
  {
    int tot = B_SZ * D_INNER;
    scan_kernel<<<(tot + 255) / 256, 256, 0, stream>>>(xc, dtb, xp, A_log, D_param, xz);
  }

  // 8) gate + cast
  {
    int tot = M_ROWS * D_INNER;
    gate_cast_kernel<<<(tot + 255) / 256, 256, 0, stream>>>(xz, yg);
  }

  // 9) out = yg @ W_out^T   (M=4096, N=1024, K=2048)
  {
    int waves = (M_ROWS / 32) * (D_MODEL / 64);  // 128 * 16 = 2048 waves
    wmma_gemm_abT<<<(waves * 32) / 256, 256, 0, stream>>>(
        yg, Wout_bf, (float*)d_out, M_ROWS, D_MODEL, D_INNER);
  }
}
